// ConvNetWordEncoder_10110353015060
// MI455X (gfx1250) — matrix operand order repaired and checked by reading, not compiled, not run on hardware
//
#include <hip/hip_runtime.h>
#include <hip/hip_bf16.h>

typedef __attribute__((ext_vector_type(4)))  __bf16 bf16x4;
typedef __attribute__((ext_vector_type(8)))  __bf16 bf16x8;
typedef __attribute__((ext_vector_type(16))) __bf16 bf16x16;
typedef __attribute__((ext_vector_type(8)))  float  f32x8;

#define HIDDEN    300
#define KSIZE     3
#define MAX_WLEN  16
#define NT_TOTAL  20                // 300 channels padded to 320 -> 20 tiles of 16
#define NTB       4                 // channel tiles per block (one per wave)
#define NGROUPS   (NT_TOTAL / NTB)  // 5
#define KTILES    10                // K padded 300 -> 320, 10 steps of 32
#define XROWS     (MAX_WLEN + 2)    // zero rows 0 and 17 = conv padding
#define XCOLS     (KTILES * 32)     // 320 (cols 300..319 stay zero)
#define PW        2                 // words processed per compute phase
#define WPB       64                // words per block (multiple of PW)
#define THREADS   (NTB * 32)        // 128

#define TILE_HALFS   (32 * 16)                            // 512 bf16 per B tile
#define GRP_HALFS    (NTB * KSIZE * KTILES * TILE_HALFS)  // 61440 bf16 = 120 KB

// ---------------------------------------------------------------------------
// Kernel 1: pack conv_w [O=300][I=300][K=3] f32 -> bf16 B-operand layout
// wp[nt][k][kk][lane][16]; B tile is 32(K) x 16(N): lane = c + 16*h,
// element e -> K = 16*h + e, N = c. Zero-padded for o>=300 / i>=300.
// ---------------------------------------------------------------------------
__global__ void pack_weights(const float* __restrict__ w,
                             __bf16* __restrict__ wp, int total) {
    int idx = blockIdx.x * blockDim.x + threadIdx.x;
    if (idx >= total) return;
    int e    = idx & 15;
    int t    = idx >> 4;
    int lane = t & 31;  t >>= 5;
    int kk   = t % KTILES; t /= KTILES;
    int k    = t % KSIZE;
    int nt   = t / KSIZE;
    int i = kk * 32 + ((lane >> 4) << 4) + e;   // input channel (K dim)
    int o = nt * 16 + (lane & 15);              // output channel (N dim)
    float v = 0.0f;
    if (o < HIDDEN && i < HIDDEN) v = w[(o * HIDDEN + i) * KSIZE + k];
    wp[idx] = (__bf16)v;
}

// ---------------------------------------------------------------------------
// Kernel 2: grid = (word chunks, 5 channel groups). Block = 4 waves; wave w
// computes channel tile grp*4+w. Weights staged once into LDS. Each phase
// gathers 2 words' embeddings; the inner loop loads each B tile from LDS once
// and applies it to both words (6 independent WMMA accumulator chains), then
// per word: max-over-time + bias + relu.
// ---------------------------------------------------------------------------
__global__ void __launch_bounds__(THREADS)
conv_encoder(const int* __restrict__ words, const float* __restrict__ emb,
             const __bf16* __restrict__ wp, const float* __restrict__ bias,
             float* __restrict__ out, int n_words) {
    __shared__ __align__(16) __bf16 Ws[GRP_HALFS];           // 120 KB
    __shared__ __align__(16) __bf16 Xs[PW][XROWS][XCOLS];    // 22.5 KB

    const int tid  = threadIdx.x;
    const int lane = tid & 31;
    const int wave = tid >> 5;          // local channel tile 0..3
    const int grp  = blockIdx.y;        // channel group 0..4
    const int h    = lane >> 4;
    const int m    = lane & 15;

    // --- Stage this group's packed weights into LDS (b128 copies). ---
    {
        const uint4* src = (const uint4*)(wp + (size_t)grp * GRP_HALFS);
        uint4* dst = (uint4*)Ws;
        for (int i = tid; i < GRP_HALFS / 8; i += THREADS) dst[i] = src[i];
    }
    // --- Zero X buffers once: pad rows/cols are never rewritten. ---
    for (int i = tid; i < PW * XROWS * XCOLS; i += THREADS)
        ((__bf16*)Xs)[i] = (__bf16)0.0f;
    __syncthreads();

    const __bf16* wbase = Ws + (size_t)wave * (KSIZE * KTILES * TILE_HALFS)
                             + lane * 16;

    for (int ph = 0; ph < WPB / PW; ++ph) {
        const int n0 = blockIdx.x * WPB + ph * PW;

        // Gather embeddings for PW words: 16 rows x 300 f32 -> bf16 LDS.
#pragma unroll
        for (int u = 0; u < PW; ++u) {
            const int n = n0 + u;
            if (n < n_words) {
                for (int idx = tid; idx < MAX_WLEN * (HIDDEN / 4); idx += THREADS) {
                    int l  = idx / (HIDDEN / 4);
                    int d4 = (idx - l * (HIDDEN / 4)) * 4;
                    int ch = words[l * n_words + n];
                    const float4 v = *(const float4*)(emb + ch * HIDDEN + d4);
                    bf16x4 p;
                    p[0] = (__bf16)v.x; p[1] = (__bf16)v.y;
                    p[2] = (__bf16)v.z; p[3] = (__bf16)v.w;
                    *(bf16x4*)&Xs[u][l + 1][d4] = p;
                }
            }
        }
        __syncthreads();

        f32x8 acc[PW][KSIZE];
#pragma unroll
        for (int u = 0; u < PW; ++u)
#pragma unroll
            for (int k = 0; k < KSIZE; ++k) acc[u][k] = (f32x8){};

#pragma unroll
        for (int kk = 0; kk < KTILES; ++kk) {
#pragma unroll
            for (int k = 0; k < KSIZE; ++k) {
                // B operand: this lane's contiguous 32 bytes, loaded once.
                const __bf16* pb = wbase + (k * KTILES + kk) * TILE_HALFS;
                bf16x8 blo = *(const bf16x8*)pb;
                bf16x8 bhi = *(const bf16x8*)(pb + 8);
                bf16x16 b = __builtin_shufflevector(
                    blo, bhi, 0,1,2,3,4,5,6,7,8,9,10,11,12,13,14,15);
#pragma unroll
                for (int u = 0; u < PW; ++u) {
                    // A operand: 16-bit 16x32 layout, row = time pos + tap.
                    const __bf16* pa = &Xs[u][m + k][kk * 32 + h * 16];
                    bf16x8 alo = *(const bf16x8*)pa;
                    bf16x8 ahi = *(const bf16x8*)(pa + 8);
                    bf16x16 a = __builtin_shufflevector(
                        alo, ahi, 0,1,2,3,4,5,6,7,8,9,10,11,12,13,14,15);
                    acc[u][k] = __builtin_amdgcn_wmma_f32_16x16x32_bf16(
                        false, a, false, b, (short)0, acc[u][k], false, false);
                }
            }
        }

        // Per word: sum tap partials, max over M (= time), bias + relu, store.
        const int o = (grp * NTB + wave) * 16 + m;
#pragma unroll
        for (int u = 0; u < PW; ++u) {
            f32x8 tot = acc[u][0] + acc[u][1] + acc[u][2];
            float mx = tot[0];
#pragma unroll
            for (int r = 1; r < 8; ++r) mx = fmaxf(mx, tot[r]);
            mx = fmaxf(mx, __shfl_xor(mx, 16, 32));
            const int n = n0 + u;
            if (n < n_words && lane < 16 && o < HIDDEN)
                out[n * HIDDEN + o] = fmaxf(mx + bias[o], 0.0f);
        }
        __syncthreads();   // protect Xs before next phase's gather
    }
}

extern "C" void kernel_launch(void* const* d_in, const int* in_sizes, int n_in,
                              void* d_out, int out_size, void* d_ws, size_t ws_size,
                              hipStream_t stream) {
    const int*   words  = (const int*)d_in[0];
    const float* emb    = (const float*)d_in[1];
    const float* conv_w = (const float*)d_in[2];
    const float* conv_b = (const float*)d_in[3];
    float*       out    = (float*)d_out;

    const int n_words = in_sizes[0] / MAX_WLEN;   // 16384

    __bf16* wp = (__bf16*)d_ws;                   // 20*3*10*512 bf16 = 600 KB
    const int total = NT_TOTAL * KSIZE * KTILES * TILE_HALFS;
    pack_weights<<<(total + 255) / 256, 256, 0, stream>>>(conv_w, wp, total);

    const int nchunks = (n_words + WPB - 1) / WPB;
    dim3 grid(nchunks, NGROUPS);
    conv_encoder<<<grid, THREADS, 0, stream>>>(words, emb, wp, conv_b,
                                               out, n_words);
}
